// LigerLMHeadKTO_46815143526662
// MI455X (gfx1250) — matrix operand order repaired and checked by reading, not compiled, not run on hardware
//
#include <hip/hip_runtime.h>
#include <math.h>

// ---------------- problem constants (from reference) ----------------
#define Bv 4
#define Tv 1024
#define Hv 1024
#define Vv 32000
#define NTOK (Bv * Tv)            // 4096 tokens
#define IGNORE_INDEX (-100)
#define BETA 0.1f

// ---------------- tiling ----------------
#define BM 64                     // tokens per workgroup (A panel resident in LDS)
#define BN 128                    // vocab columns per chunk (8 waves x 16 cols)
#define NWAVE 8
#define BLOCK (NWAVE * 32)        // 256 threads = 8 wave32
#define NCHUNK (Vv / BN)          // 250
#define KSTEP 32                  // K per bf16 WMMA
#define NKS (Hv / KSTEP)          // 32 k-steps
#define KB (Hv / 8)               // 128 8-wide K blocks (LDS tiling granule)
// LDS layout, hi/lo interleaved per K block (all DS offsets fold into 16-bit imm):
//   [kb 0..127] { hi: 4 tiles of [16 rows x 8 bf16] (1 KB) | lo: 4 tiles (1 KB) }
//   ushort index: kb*1024 + (lo?512:0) + rt*128 + row*8 + j ; total = 256 KB
// Global W split layout, hi/lo interleaved per column (lo = +2048 B global imm):
//   [col 0..V) { hi: H bf16 | lo: H bf16 }

typedef unsigned int   uint_t;
typedef unsigned short ushort_t;
typedef __attribute__((ext_vector_type(8)))  float  v8f;
typedef __attribute__((ext_vector_type(16))) __bf16 v16bf;

union BV { v16bf v; uint4 q[2]; };   // 8 VGPRs = 16 bf16

// split f32 -> bf16 hi (RNE) + bf16 lo (RNE of residual); a ~= hi + lo, err ~2^-16
__device__ __forceinline__ void bsplit(float f, uint_t& h, uint_t& l) {
    uint_t u  = __float_as_uint(f);
    uint_t hr = ((u + 0x7FFFu + ((u >> 16) & 1u)) >> 16) & 0xFFFFu;
    float  r  = f - __uint_as_float(hr << 16);
    uint_t u2 = __float_as_uint(r);
    h = hr;
    l = ((u2 + 0x7FFFu + ((u2 >> 16) & 1u)) >> 16) & 0xFFFFu;
}

// ---------------- one-time W split: f32 [V,H] -> interleaved bf16 hi/lo ----------------
__global__ void __launch_bounds__(256)
wsplit_kernel(const float* __restrict__ src, ushort_t* __restrict__ dst, int n4) {
    const int i = blockIdx.x * 256 + threadIdx.x;    // float4 index over [V,H]
    if (i >= n4) return;
    const float4 f = ((const float4*)src)[i];
    uint_t h0, h1, h2, h3, l0, l1, l2, l3;
    bsplit(f.x, h0, l0); bsplit(f.y, h1, l1);
    bsplit(f.z, h2, l2); bsplit(f.w, h3, l3);
    ushort4 hv; hv.x = (ushort_t)h0; hv.y = (ushort_t)h1; hv.z = (ushort_t)h2; hv.w = (ushort_t)h3;
    ushort4 lv; lv.x = (ushort_t)l0; lv.y = (ushort_t)l1; lv.z = (ushort_t)l2; lv.w = (ushort_t)l3;
    const int col = i >> 8;                          // H/4 = 256 float4 per column
    const int h4  = i & 255;
    ((ushort4*)dst)[(size_t)col * 512 + h4]       = hv;   // hi half of column
    ((ushort4*)dst)[(size_t)col * 512 + 256 + h4] = lv;   // lo half of column
}

// ---------------- fused split-bf16 GEMM + online logsumexp + target logit ----------------
// D(16x16 f32) += A(16x32 bf16) * B(32x16 bf16), wave32.
// A layout: lanes 0-15 -> M=lane, K {0..7,16..23}; lanes 16-31 -> M=lane-16, K {8..15,24..31}
// B layout: lanes 0-15 -> N=lane, K {0..7,16..23}; lanes 16-31 -> N=lane-16, K {8..15,24..31}
// C/D:      VGPR i, lanes 0-15 -> row i, lanes 16-31 -> row i+8; col = lane&15
__global__ void __launch_bounds__(BLOCK)
kto_gemm_lse_kernel(const float* __restrict__ x,        // [NTOK,H] f32 policy hidden
                    const float* __restrict__ ref_x,    // [NTOK,H] f32 ref hidden
                    const ushort_t* __restrict__ w2,    // [V][2H] bf16 hi|lo policy head
                    const ushort_t* __restrict__ rw2,   // [V][2H] bf16 hi|lo ref head
                    const int* __restrict__ y,          // [NTOK]
                    float* __restrict__ ws)             // [4*NTOK] lse/tgt x 2 models
{
    extern __shared__ ushort_t AT[];                    // interleaved hi/lo panel, 256 KB
    __shared__ float partM[NWAVE][BM];
    __shared__ float partS[NWAVE][BM];
    __shared__ float runM[BM], runS[BM], tgtL[BM];
    __shared__ int   yL[BM];

    const int model = blockIdx.y;                       // 0 = policy, 1 = reference
    const float*    __restrict__ X  = model ? ref_x : x;
    const ushort_t* __restrict__ Wm = model ? rw2 : w2;
    float* __restrict__ lse_out = ws + (size_t)model * (2 * NTOK);
    float* __restrict__ tgt_out = lse_out + NTOK;

    const int tid  = threadIdx.x;
    const int lane = tid & 31;
    const int wave = tid >> 5;
    const int l16  = lane & 15;
    const int s    = lane >> 4;                         // K-half select
    const int tokbase = blockIdx.x * BM;

    // ---- stage + split A panel (BM x H f32 -> interleaved bf16 hi/lo LDS tiles) ----
    for (int i = tid; i < BM * KB; i += BLOCK) {
        const int r  = i >> 7;                          // / KB
        const int c8 = i & (KB - 1);
        const float* src = X + (size_t)(tokbase + r) * Hv + c8 * 8;
        const float4 f0 = ((const float4*)src)[0];
        const float4 f1 = ((const float4*)src)[1];
        const float  f[8] = {f0.x, f0.y, f0.z, f0.w, f1.x, f1.y, f1.z, f1.w};
        uint_t h[8], l[8];
        #pragma unroll
        for (int j = 0; j < 8; ++j) bsplit(f[j], h[j], l[j]);
        uint4 qh, ql;
        qh.x = h[0] | (h[1] << 16); qh.y = h[2] | (h[3] << 16);
        qh.z = h[4] | (h[5] << 16); qh.w = h[6] | (h[7] << 16);
        ql.x = l[0] | (l[1] << 16); ql.y = l[2] | (l[3] << 16);
        ql.z = l[4] | (l[5] << 16); ql.w = l[6] | (l[7] << 16);
        const int tb = c8 * 1024 + (r >> 4) * 128 + (r & 15) * 8;
        *(uint4*)&AT[tb]       = qh;                    // hi tile group
        *(uint4*)&AT[tb + 512] = ql;                    // lo tile group
    }
    if (tid < BM) {
        runM[tid] = -__builtin_inff();
        runS[tid] = 0.0f;
        tgtL[tid] = 0.0f;
        yL[tid]   = y[tokbase + tid];
    }
    __syncthreads();

    const int abase0 = s * 1024 + l16 * 8;              // per-lane ushort base in a kk group

    for (int c = 0; c < NCHUNK; ++c) {
        const int col0 = c * BN + wave * 16;            // this wave's 16 columns
        // one pointer per lane; lo plane is +Hv ushorts (+2048 B) -> global imm offset
        const ushort_t* bp0 = Wm + (size_t)(col0 + l16) * (2 * Hv) + s * 8;
        if (c + 1 < NCHUNK)                             // warm next W slab (global_prefetch_b8)
            __builtin_prefetch(bp0 + (size_t)BN * (2 * Hv), 0, 1);

        v8f acc[4] = {v8f{}, v8f{}, v8f{}, v8f{}};
        #pragma unroll 2
        for (int kk = 0; kk < NKS; ++kk) {
            const ushort_t* bp = bp0 + kk * KSTEP;
            BV bh, bl;
            bh.q[0] = *(const uint4*)(bp);
            bh.q[1] = *(const uint4*)(bp + 16);
            bl.q[0] = *(const uint4*)(bp + Hv);         // +2048 B immediate
            bl.q[1] = *(const uint4*)(bp + Hv + 16);

            const int ab = kk * 4096 + abase0;          // one LDS base, imm offsets below
            #pragma unroll
            for (int rt = 0; rt < 4; ++rt) {
                BV ah, al;
                ah.q[0] = *(const uint4*)&AT[ab + rt * 128];           // kb,   hi
                ah.q[1] = *(const uint4*)&AT[ab + rt * 128 + 2048];    // kb+2, hi
                al.q[0] = *(const uint4*)&AT[ab + rt * 128 + 512];     // kb,   lo
                al.q[1] = *(const uint4*)&AT[ab + rt * 128 + 2560];    // kb+2, lo
                // split product: hi*hi + hi*lo + lo*hi  (~f32 accuracy)
                acc[rt] = __builtin_amdgcn_wmma_f32_16x16x32_bf16(false, ah.v, false, bh.v, (short)0, acc[rt], false, false);
                acc[rt] = __builtin_amdgcn_wmma_f32_16x16x32_bf16(false, ah.v, false, bl.v, (short)0, acc[rt], false, false);
                acc[rt] = __builtin_amdgcn_wmma_f32_16x16x32_bf16(false, al.v, false, bh.v, (short)0, acc[rt], false, false);
            }
        }

        // ---- epilogue: per-row max / sum-exp over this wave's 16 columns ----
        const int colMine = col0 + l16;
        #pragma unroll
        for (int rt = 0; rt < 4; ++rt) {
            const v8f a = acc[rt];
            #pragma unroll
            for (int i = 0; i < 8; ++i) {
                const float v = a[i];
                float m = v;
                m = fmaxf(m, __shfl_xor(m, 1, 32));
                m = fmaxf(m, __shfl_xor(m, 2, 32));
                m = fmaxf(m, __shfl_xor(m, 4, 32));
                m = fmaxf(m, __shfl_xor(m, 8, 32));
                float e = __expf(v - m);
                e += __shfl_xor(e, 1, 32);
                e += __shfl_xor(e, 2, 32);
                e += __shfl_xor(e, 4, 32);
                e += __shfl_xor(e, 8, 32);
                const int rowl = rt * 16 + i + (s << 3);
                if (l16 == 0) { partM[wave][rowl] = m; partS[wave][rowl] = e; }
                if (colMine == yL[rowl]) tgtL[rowl] = v;   // unique lane over whole V loop
            }
        }
        __syncthreads();
        if (tid < BM) {                                    // online logsumexp merge
            float M = runM[tid], S = runS[tid];
            #pragma unroll
            for (int w = 0; w < NWAVE; ++w) {
                const float m2 = partM[w][tid], s2 = partS[w][tid];
                const float nm = fmaxf(M, m2);
                S = S * __expf(M - nm) + s2 * __expf(m2 - nm);
                M = nm;
            }
            runM[tid] = M; runS[tid] = S;
        }
        __syncthreads();
    }

    if (tid < BM) {
        lse_out[tokbase + tid] = runM[tid] + __logf(runS[tid]);
        tgt_out[tokbase + tid] = tgtL[tid];
    }
}

// ---------------- final KTO loss reduction (tiny) ----------------
__global__ void __launch_bounds__(256)
kto_finalize_kernel(const float* __restrict__ ws,
                    const int* __restrict__ y,
                    const unsigned char* __restrict__ pref,   // bool per batch
                    float* __restrict__ out)
{
    __shared__ float red[256];
    __shared__ float totals[12];
    const int tid = threadIdx.x;

    float pol[Bv] = {0, 0, 0, 0}, ref[Bv] = {0, 0, 0, 0}, cnt[Bv] = {0, 0, 0, 0};
    for (int t = tid; t < NTOK; t += 256) {
        const int b = t >> 10;                      // T = 1024
        const float mask = (y[t] != IGNORE_INDEX) ? 1.0f : 0.0f;
        pol[b] += mask * (ws[NTOK + t]     - ws[t]);             // tgt - lse (policy)
        ref[b] += mask * (ws[3 * NTOK + t] - ws[2 * NTOK + t]);  // tgt - lse (reference)
        cnt[b] += mask;
    }
    float vals[12] = {pol[0], pol[1], pol[2], pol[3],
                      ref[0], ref[1], ref[2], ref[3],
                      cnt[0], cnt[1], cnt[2], cnt[3]};
    for (int k = 0; k < 12; ++k) {
        red[tid] = vals[k];
        __syncthreads();
        for (int st = 128; st > 0; st >>= 1) {
            if (tid < st) red[tid] += red[tid + st];
            __syncthreads();
        }
        if (tid == 0) totals[k] = red[0];
        __syncthreads();
    }
    if (tid == 0) {
        float loss = 0.0f;
        #pragma unroll
        for (int b = 0; b < Bv; ++b) {
            const float inv = 1.0f / totals[8 + b];
            const float lr = totals[b] * inv - totals[4 + b] * inv;   // logratios
            const float chosen   = 1.0f - 1.0f / (1.0f + __expf(-BETA * lr));
            const float rejected = 1.0f - 1.0f / (1.0f + __expf( BETA * lr));
            loss += pref[b] ? chosen : rejected;
        }
        out[0] = loss * (1.0f / Bv);
    }
}

extern "C" void kernel_launch(void* const* d_in, const int* in_sizes, int n_in,
                              void* d_out, int out_size, void* d_ws, size_t ws_size,
                              hipStream_t stream) {
    (void)in_sizes; (void)n_in; (void)out_size; (void)ws_size;
    const float* x     = (const float*)d_in[0];   // [B,T,H] f32
    const float* ref_x = (const float*)d_in[1];   // [B,T,H] f32
    const int*   y     = (const int*)d_in[2];     // [B,T] int
    const unsigned char* pref = (const unsigned char*)d_in[3]; // [B] bool
    const float* W     = (const float*)d_in[4];   // [V,H] f32
    const float* ref_W = (const float*)d_in[5];   // [V,H] f32
    float* out = (float*)d_out;

    // workspace layout (assumes ws_size >= ~263 MB):
    //   [0, 4*NTOK) f32 : lse/tgt for policy+reference
    //   then 2 interleaved bf16 hi|lo planes of 2*V*H ushorts: w2, rw2
    float* ws = (float*)d_ws;
    ushort_t* w2  = (ushort_t*)(ws + 4 * NTOK);
    ushort_t* rw2 = w2 + (size_t)2 * Vv * Hv;

    // 1) one-time split of the heads into interleaved bf16 hi/lo (memory-bound, ~17 us)
    const int n4 = (Vv * Hv) / 4;
    wsplit_kernel<<<n4 / 256, 256, 0, stream>>>(W,     w2,  n4);
    wsplit_kernel<<<n4 / 256, 256, 0, stream>>>(ref_W, rw2, n4);

    // 2) fused split-bf16 GEMM + logsumexp, both models concurrently
    const size_t shbytes = 256 * 1024;            // interleaved A panel
    hipFuncSetAttribute((const void*)kto_gemm_lse_kernel,
                        hipFuncAttributeMaxDynamicSharedMemorySize, (int)shbytes);
    dim3 grid(NTOK / BM, 2);                      // 64 token tiles x {policy, reference}
    kto_gemm_lse_kernel<<<grid, BLOCK, shbytes, stream>>>(x, ref_x, w2, rw2, y, ws);

    // 3) tiny KTO reduction
    kto_finalize_kernel<<<1, 256, 0, stream>>>(ws, y, pref, out);
}
